// MultiHeadAttention_85091892068655
// MI455X (gfx1250) — matrix operand order repaired and checked by reading, not compiled, not run on hardware
//
#include <hip/hip_runtime.h>
#include <hip/hip_bf16.h>

#define NUM_HEADS 16
#define HEAD_SIZE 64
#define N_EMBD    1024
#define BATCH     4
#define SEQ       2048
#define M_TOK     (BATCH*SEQ)      // 8192 tokens
#define QKV_N     (3*N_EMBD)       // 3072

typedef __attribute__((ext_vector_type(16))) __bf16 v16bf;
typedef __attribute__((ext_vector_type(8)))  __bf16 v8bf;
typedef __attribute__((ext_vector_type(8)))  float  v8f;

__device__ __forceinline__ v8f wmma_bf16(v16bf a, v16bf b, v8f c) {
  // (neg_a, A, neg_b, B, c_mod, C, reuse_a, reuse_b)
  return __builtin_amdgcn_wmma_f32_16x16x32_bf16(false, a, false, b, (short)0, c, false, false);
}

// Cross-half lane swap (lane i <-> lane i^16) as a pure VALU op:
// v_permlanex16_b32 with identity lane selects. Avoids ds_bpermute + DScnt
// waits in the softmax critical path.
__device__ __forceinline__ float half_swap(float v) {
#if __has_builtin(__builtin_amdgcn_permlanex16)
  int i = __builtin_bit_cast(int, v);
  i = __builtin_amdgcn_permlanex16(i, i, 0x76543210, 0xFEDCBA98, false, false);
  return __builtin_bit_cast(float, i);
#else
  return __shfl_xor(v, 16, 32);
#endif
}

// A-fragment (16x32 bf16): lane&15 = row M.
// lanes 0-15 : K = {0..15} in elems 0-15
// lanes 16-31: K = {16..31} in elems 0-15
// kb = 0 (lanes<16) or 8 (lanes>=16)
__device__ __forceinline__ v16bf load_fragA(const __bf16* row, int kb) {
  v8bf lo = *(const v8bf*)(row + 2 * kb);
  v8bf hi = *(const v8bf*)(row + 2 * kb + 8);
  v16bf r;
#pragma unroll
  for (int i = 0; i < 8; ++i) { r[i] = lo[i]; r[i + 8] = hi[i]; }
  return r;
}

// B-fragment (32x16 bf16): lane&15 = col N.
// lanes 0-15 hold K=0..15 contiguously, lanes 16-31 hold K=16..31.
// hi16 = 0 (lanes<16) or 16 (lanes>=16)
__device__ __forceinline__ v16bf load_fragB(const __bf16* col, int hi16) {
  const __bf16* p = col + hi16;
  v8bf lo = *(const v8bf*)(p);
  v8bf hi = *(const v8bf*)(p + 8);
  v16bf r;
#pragma unroll
  for (int i = 0; i < 8; ++i) { r[i] = lo[i]; r[i + 8] = hi[i]; }
  return r;
}

// ---------------- conversion kernels ----------------

__global__ void k_cvt_bf16(const float* __restrict__ src, __bf16* __restrict__ dst, int n) {
  int i = blockIdx.x * blockDim.x + threadIdx.x;
  if (i < n) dst[i] = (__bf16)src[i];
}

// w [K,N] fp32 row-major -> wt [N,K] bf16 row-major (so B-fragments load contiguously)
__global__ void k_transpose_bf16(const float* __restrict__ w, __bf16* __restrict__ wt, int K, int N) {
  int i = blockIdx.x * blockDim.x + threadIdx.x;
  if (i >= K * N) return;
  int k = i / N;
  int n = i - k * N;
  wt[(size_t)n * K + k] = (__bf16)w[i];
}

// ---------------- WMMA GEMM: C[M,N] = A[M,K] * Bt[N,K]^T (+bias) ----------------
// One wave computes a 32x64 output tile: 2 A-fragments x 4 B-fragments,
// 8 accumulators; each B (weight) fragment load feeds 2 WMMAs.

template <bool OUT_BF16, bool ADD_BIAS>
__global__ __launch_bounds__(256, 1)
void k_gemm_wmma(const __bf16* __restrict__ A, const __bf16* __restrict__ Bt,
                 void* __restrict__ Cout, const float* __restrict__ bias,
                 int M, int N, int K) {
  const int lane = threadIdx.x & 31;
  const int wave = blockIdx.x * (blockDim.x >> 5) + (threadIdx.x >> 5);
  const int nT = N >> 6;                 // tiles of 64 in N
  const int mt = wave / nT;              // tiles of 32 in M
  const int nt = wave - mt * nT;
  if (mt * 32 >= M) return;
  const int lp  = lane & 15;
  const int kbA = (lane < 16) ? 0 : 8;
  const int hiB = (lane < 16) ? 0 : 16;
  const __bf16* Arow0 = A + (size_t)(mt * 32 + lp) * K;
  const __bf16* Arow1 = Arow0 + (size_t)16 * K;
  const __bf16* Brow[4];
#pragma unroll
  for (int j = 0; j < 4; ++j)
    Brow[j] = Bt + (size_t)(nt * 64 + j * 16 + lp) * K;

  v8f acc[2][4] = {};
  for (int k0 = 0; k0 < K; k0 += 32) {
    __builtin_prefetch((const void*)(Arow0 + k0 + 128), 0, 0);
    v16bf a0 = load_fragA(Arow0 + k0, kbA);
    v16bf a1 = load_fragA(Arow1 + k0, kbA);
#pragma unroll
    for (int j = 0; j < 4; ++j) {
      v16bf b = load_fragB(Brow[j] + k0, hiB);
      acc[0][j] = wmma_bf16(a0, b, acc[0][j]);
      acc[1][j] = wmma_bf16(a1, b, acc[1][j]);
    }
  }
  // D layout: lanes 0-15 -> N=lane, M=vgpr r; lanes 16-31 -> N=lane-16, M=r+8
  const int mb = (lane < 16) ? 0 : 8;
#pragma unroll
  for (int i = 0; i < 2; ++i) {
#pragma unroll
    for (int j = 0; j < 4; ++j) {
      const int n = nt * 64 + j * 16 + lp;
      const float bv = ADD_BIAS ? bias[n] : 0.0f;
#pragma unroll
      for (int r = 0; r < 8; ++r) {
        const size_t idx = (size_t)(mt * 32 + i * 16 + mb + r) * N + n;
        const float v = acc[i][j][r] + bv;
        if (OUT_BF16) ((__bf16*)Cout)[idx] = (__bf16)v;
        else          ((float*)Cout)[idx]  = v;
      }
    }
  }
}

// ---------------- RoPE + head split ----------------
// qkv bf16 -> Q (pre-scaled by 1/sqrt(hs)=0.125, exact in bf16), K [bh,T,hs],
// V^T [bh,hs,T]

__global__ void k_rope_split(const __bf16* __restrict__ qkv,
                             __bf16* __restrict__ Q, __bf16* __restrict__ Kc,
                             __bf16* __restrict__ Vt) {
  int idx = blockIdx.x * blockDim.x + threadIdx.x;
  if (idx >= BATCH * SEQ * NUM_HEADS) return;
  const int h = idx % NUM_HEADS;
  const int t = (idx / NUM_HEADS) % SEQ;
  const int b = idx / (NUM_HEADS * SEQ);
  const __bf16* row = qkv + (size_t)(b * SEQ + t) * QKV_N + h * HEAD_SIZE;
  const size_t bh = (size_t)b * NUM_HEADS + h;
  __bf16* qo = Q  + (bh * SEQ + t) * HEAD_SIZE;
  __bf16* ko = Kc + (bh * SEQ + t) * HEAD_SIZE;
  __bf16* vo = Vt + bh * HEAD_SIZE * SEQ + t;
#pragma unroll
  for (int i = 0; i < HEAD_SIZE / 2; ++i) {
    const float inv = __powf(10000.0f, -(float)(2 * i) * (1.0f / HEAD_SIZE));
    float s, c;
    __sincosf((float)t * inv, &s, &c);
    const float q0 = (float)row[2 * i], q1 = (float)row[2 * i + 1];
    const float k0 = (float)row[N_EMBD + 2 * i], k1 = (float)row[N_EMBD + 2 * i + 1];
    qo[2 * i]     = (__bf16)((q0 * c - q1 * s) * 0.125f);   // fold softmax scale into Q
    qo[2 * i + 1] = (__bf16)((q0 * s + q1 * c) * 0.125f);
    ko[2 * i]     = (__bf16)(k0 * c - k1 * s);
    ko[2 * i + 1] = (__bf16)(k0 * s + k1 * c);
    vo[(2 * i) * SEQ]     = row[2 * N_EMBD + 2 * i];
    vo[(2 * i + 1) * SEQ] = row[2 * N_EMBD + 2 * i + 1];
  }
}

// ---------------- causal flash attention, all-WMMA, S^T formulation ----------------
// One wave owns (b,h, 32-query tile): two 16-query subtiles share all K/V
// fragment loads. S^T = K_tile(A) x Q^T(B): D lanes = query columns, so softmax
// reductions are in-lane (+one permlanex16 across the key-half split), flash
// rescale is a per-lane scalar, and P^T re-packs into the PV B-fragment with
// in-lane bf16 packing + one lane-pair exchange. Only the single diagonal block
// takes the masked (compare/select) path; steady-state blocks are mask-free.
// Only 2 K-fragments are live at any time (hs-half loop) to stay spill-free.

template <bool MASK>
__device__ __forceinline__ void attn_block(
    int ks, int lane, int lp, int kbA, int mb,
    const int (&qidx)[2],
    const __bf16* __restrict__ Kbase, const __bf16* __restrict__ Vbase,
    const v16bf (&qB)[2][2], v8f (&acc)[2][4],
    float (&mrun)[2], float (&lrun)[2]) {
  // S^T tiles: st[s][keyhalf]; accumulate over the two hs halves so that only
  // two K-fragments are live concurrently.
  v8f st[2][2] = {};
#pragma unroll
  for (int hh = 0; hh < 2; ++hh) {
    v16bf kf0 = load_fragA(Kbase + (size_t)(ks + lp) * HEAD_SIZE + hh * 32, kbA);
    v16bf kf1 = load_fragA(Kbase + (size_t)(ks + 16 + lp) * HEAD_SIZE + hh * 32, kbA);
#pragma unroll
    for (int s = 0; s < 2; ++s) {
      st[s][0] = wmma_bf16(kf0, qB[s][hh], st[s][0]);
      st[s][1] = wmma_bf16(kf1, qB[s][hh], st[s][1]);
    }
  }

  v16bf pB[2];
#pragma unroll
  for (int s = 0; s < 2; ++s) {
    float s0[8], s1[8];
    float bmax = -1e30f;
#pragma unroll
    for (int r = 0; r < 8; ++r) {
      float v0 = st[s][0][r];
      float v1 = st[s][1][r];
      if (MASK) {
        if (ks + mb + r > qidx[s])      v0 = -1e30f;
        if (ks + 16 + mb + r > qidx[s]) v1 = -1e30f;
      }
      s0[r] = v0; s1[r] = v1;
      bmax = fmaxf(bmax, fmaxf(v0, v1));
    }
    bmax = fmaxf(bmax, half_swap(bmax)); // combine key halves (lane n <-> n+16)
    const float mnew  = fmaxf(mrun[s], bmax);
    const float alpha = __expf(mrun[s] - mnew);
    float psum = 0.0f;
#pragma unroll
    for (int r = 0; r < 8; ++r) {
      const float p0 = __expf(s0[r] - mnew);
      const float p1 = __expf(s1[r] - mnew);
      psum += p0 + p1;
      // pack P^T into B-fragment: lanes<16 need keys 0..15, lanes>=16 need 16..31
      const float x0 = half_swap(p0);
      const float x1 = half_swap(p1);
      if (lane < 16) { pB[s][r] = (__bf16)p0; pB[s][r + 8] = (__bf16)x0; }
      else           { pB[s][r] = (__bf16)x1; pB[s][r + 8] = (__bf16)p1; }
    }
    psum += half_swap(psum);
    lrun[s] = lrun[s] * alpha + psum;
    mrun[s] = mnew;
#pragma unroll
    for (int dt = 0; dt < 4; ++dt)
#pragma unroll
      for (int r = 0; r < 8; ++r) acc[s][dt][r] *= alpha;
  }
  // y^T += V^T(A: 16d x 32k) * P^T(B: 32k x 16q); V fragment shared by both subtiles
#pragma unroll
  for (int dt = 0; dt < 4; ++dt) {
    v16bf vf = load_fragA(Vbase + (size_t)(dt * 16 + lp) * SEQ + ks, kbA);
    acc[0][dt] = wmma_bf16(vf, pB[0], acc[0][dt]);
    acc[1][dt] = wmma_bf16(vf, pB[1], acc[1][dt]);
  }
}

__global__ __launch_bounds__(256, 1)
void k_attention_wmma(const __bf16* __restrict__ Q, const __bf16* __restrict__ Kc,
                      const __bf16* __restrict__ Vt, __bf16* __restrict__ Y) {
  const int lane = threadIdx.x & 31;
  const int wave = blockIdx.x * (blockDim.x >> 5) + (threadIdx.x >> 5);
  const int nqt = SEQ / 32;              // 64 query tiles of 32
  const int qt = wave % nqt;
  const int bh = wave / nqt;
  if (bh >= BATCH * NUM_HEADS) return;
  const int lp  = lane & 15;
  const int kbA = (lane < 16) ? 0 : 8;
  const int hiB = (lane < 16) ? 0 : 16;
  const int mb  = kbA;                   // D-row split is the same 0/8
  const int q0base = qt * 32;

  v16bf qB[2][2];
#pragma unroll
  for (int s = 0; s < 2; ++s) {
    const __bf16* Qrow = Q + ((size_t)bh * SEQ + q0base + s * 16 + lp) * HEAD_SIZE;
    qB[s][0] = load_fragB(Qrow, hiB);       // hs dims 0..31
    qB[s][1] = load_fragB(Qrow + 32, hiB);  // hs dims 32..63
  }
  const __bf16* Kbase = Kc + (size_t)bh * SEQ * HEAD_SIZE;
  const __bf16* Vbase = Vt + (size_t)bh * HEAD_SIZE * SEQ;

  v8f acc[2][4] = {};
  float mrun[2] = {-1e30f, -1e30f};
  float lrun[2] = {0.0f, 0.0f};
  const int qidx[2] = {q0base + lp, q0base + 16 + lp};

  // full (unmasked) blocks strictly below the diagonal
  for (int blk = 0; blk < qt; ++blk) {
    __builtin_prefetch((const void*)(Kbase + (size_t)(blk * 32 + 32 + lp) * HEAD_SIZE), 0, 0);
    attn_block<false>(blk * 32, lane, lp, kbA, mb, qidx, Kbase, Vbase,
                      qB, acc, mrun, lrun);
  }
  // single masked diagonal block (keys q0base .. q0base+31)
  attn_block<true>(qt * 32, lane, lp, kbA, mb, qidx, Kbase, Vbase,
                   qB, acc, mrun, lrun);

  const int b = bh / NUM_HEADS, h = bh % NUM_HEADS;
#pragma unroll
  for (int s = 0; s < 2; ++s) {
    const float inv = 1.0f / lrun[s];
    const size_t orow =
        (size_t)(b * SEQ + q0base + s * 16 + lp) * N_EMBD + h * HEAD_SIZE;
#pragma unroll
    for (int dt = 0; dt < 4; ++dt)
#pragma unroll
      for (int r = 0; r < 8; ++r)
        Y[orow + dt * 16 + mb + r] = (__bf16)(acc[s][dt][r] * inv);
  }
}

// ---------------- host launcher ----------------

extern "C" void kernel_launch(void* const* d_in, const int* in_sizes, int n_in,
                              void* d_out, int out_size, void* d_ws, size_t ws_size,
                              hipStream_t stream) {
  const float* x      = (const float*)d_in[0];
  const float* W_qkv  = (const float*)d_in[1];
  const float* W_proj = (const float*)d_in[2];
  const float* b_proj = (const float*)d_in[3];

  char* ws = (char*)d_ws;
  size_t off = 0;
  auto take = [&](size_t bytes) -> void* {
    void* p = ws + off;
    off += (bytes + 255) & ~(size_t)255;
    return p;
  };
  __bf16* Xb   = (__bf16*)take((size_t)M_TOK * N_EMBD * 2);   // x in bf16
  __bf16* WqT  = (__bf16*)take((size_t)QKV_N * N_EMBD * 2);   // W_qkv^T bf16
  __bf16* WpT  = (__bf16*)take((size_t)N_EMBD * N_EMBD * 2);  // W_proj^T bf16
  __bf16* QKVb = (__bf16*)take((size_t)M_TOK * QKV_N * 2);    // qkv bf16
  __bf16* Qb   = (__bf16*)take((size_t)M_TOK * N_EMBD * 2);   // [bh,T,hs], pre-scaled
  __bf16* Kb   = (__bf16*)take((size_t)M_TOK * N_EMBD * 2);   // [bh,T,hs]
  __bf16* Vt   = (__bf16*)take((size_t)M_TOK * N_EMBD * 2);   // [bh,hs,T]
  __bf16* Yb   = (__bf16*)take((size_t)M_TOK * N_EMBD * 2);   // attn out bf16
  (void)ws_size; (void)in_sizes; (void)n_in; (void)out_size;

  {
    int n = M_TOK * N_EMBD;
    k_cvt_bf16<<<(n + 255) / 256, 256, 0, stream>>>(x, Xb, n);
  }
  {
    int n = N_EMBD * QKV_N;
    k_transpose_bf16<<<(n + 255) / 256, 256, 0, stream>>>(W_qkv, WqT, N_EMBD, QKV_N);
  }
  {
    int n = N_EMBD * N_EMBD;
    k_transpose_bf16<<<(n + 255) / 256, 256, 0, stream>>>(W_proj, WpT, N_EMBD, N_EMBD);
  }
  {
    int waves = (M_TOK / 32) * (QKV_N / 64);    // 12288 waves, 8 waves/block
    k_gemm_wmma<true, false><<<waves / 8, 256, 0, stream>>>(
        Xb, WqT, (void*)QKVb, nullptr, M_TOK, QKV_N, N_EMBD);
  }
  {
    int n = BATCH * SEQ * NUM_HEADS;
    k_rope_split<<<(n + 255) / 256, 256, 0, stream>>>(QKVb, Qb, Kb, Vt);
  }
  {
    int waves = BATCH * NUM_HEADS * (SEQ / 32); // 4096 waves
    k_attention_wmma<<<waves / 8, 256, 0, stream>>>(Qb, Kb, Vt, Yb);
  }
  {
    int waves = (M_TOK / 32) * (N_EMBD / 64);   // 4096 waves
    k_gemm_wmma<false, true><<<waves / 8, 256, 0, stream>>>(
        Yb, WpT, d_out, b_proj, M_TOK, N_EMBD, N_EMBD);
  }
}
